// MAEDecoder_89069031784576
// MI455X (gfx1250) — matrix operand order repaired and checked by reading, not compiled, not run on hardware
//
#include <hip/hip_runtime.h>
#include <hip/hip_bf16.h>

// ---- problem constants (from reference) ----
#define T_TOK   6272      // B * NUM_PATCHES = 32 * 196
#define D_EMB   512
#define FF_DIM  2048
#define NLAYERS 8
#define NHEADS  16
#define HD      32
#define NPAT    196
#define BSZ     32
#define NVIS    49
#define NKPAD   224       // 196 keys padded to 13*16=208, then to 224 (7 chunks of 32)

typedef unsigned short u16;
typedef unsigned int   u32;
typedef __attribute__((ext_vector_type(16))) __bf16 v16bf;
typedef __attribute__((ext_vector_type(8)))  float  v8f;
typedef int v4i_async __attribute__((vector_size(16)));

union FragU { uint4 q[2]; v16bf v; };

__device__ __forceinline__ u16 f2bf(float f) {
  u32 u = __float_as_uint(f);
  u32 r = u + 0x7FFFu + ((u >> 16) & 1u);   // round-to-nearest-even
  return (u16)(r >> 16);
}

__device__ __forceinline__ v8f wmma_bf16(const FragU& a, const FragU& b, v8f c) {
  return __builtin_amdgcn_wmma_f32_16x16x32_bf16(false, a.v, false, b.v, (short)0, c, false, false);
}

// ---- CDNA5 async global->LDS copy (ASYNCcnt path); falls back to ld+st ----
__device__ __forceinline__ void asyncCopyB128(const u16* g, u16* l) {
#if __has_builtin(__builtin_amdgcn_global_load_async_to_lds_b128)
  __builtin_amdgcn_global_load_async_to_lds_b128(
      (__attribute__((address_space(1))) v4i_async*)(size_t)g,
      (__attribute__((address_space(3))) v4i_async*)(u32)(size_t)l,
      0, 0);
#else
  *(uint4*)l = *(const uint4*)g;
#endif
}
__device__ __forceinline__ void waitAsync() {
#if __has_builtin(__builtin_amdgcn_global_load_async_to_lds_b128)
  asm volatile("s_wait_asynccnt 0x0" ::: "memory");
#endif
}

// ---------------- f32 -> bf16 convert ----------------
__global__ void k_f32_to_bf16(const float* __restrict__ in, u16* __restrict__ out, int n) {
  int i = blockIdx.x * blockDim.x + threadIdx.x;
  if (i < n) out[i] = f2bf(in[i]);
}

// ---------------- token gather + pos-emb ----------------
__global__ __launch_bounds__(512) void k_embed(const float* __restrict__ xe,
                                               const int* __restrict__ idx,
                                               const float* __restrict__ mtok,
                                               const float* __restrict__ pos,
                                               float* __restrict__ x) {
  int bp = blockIdx.x;                 // 0 .. T_TOK-1
  int b = bp / NPAT, p = bp % NPAT;
  int d = threadIdx.x;                 // 0 .. 511
  int j = idx[b * NPAT + p];
  float v = (j < NVIS) ? xe[((size_t)b * (NVIS + 1) + j + 1) * D_EMB + d] : mtok[d];
  x[(size_t)bp * D_EMB + d] = v + pos[(size_t)p * D_EMB + d];
}

// ---------------- WMMA bf16 GEMM (double-buffered LDS + async prefetch) ----------------
// C[M,N] = (A[M,K] @ B[K,N] + bias) * scale ; RELU/OUTBF are compile-time.
// Block tile 128x64, BK=32, 256 threads = 8 waves (4 along M x 2 along N),
// each wave a 32x32 tile = 2x2 v_wmma_f32_16x16x32_bf16 accumulators.
// Requires M%128==0, N%64==0, K%32==0 (true for all calls here).
template <bool OUTBF, bool RELU>
__global__ __launch_bounds__(256) void k_gemm_bf16(
    const u16* __restrict__ A, const u16* __restrict__ B,
    const float* __restrict__ bias,
    float* __restrict__ outF, u16* __restrict__ outB,
    int M, int N, int K, float scale)
{
  __shared__ __align__(16) u16 As[2][128][40];   // [buf][m][k], padded stride 40
  __shared__ __align__(16) u16 Bs[2][64][40];    // [buf][n][k] (transposed), padded

  const int tid  = threadIdx.x;
  const int lane = tid & 31;
  const int wave = tid >> 5;
  const int mOff = (wave & 3) * 32;
  const int nOff = (wave >> 2) * 32;
  const int rowBase = blockIdx.x * 128;
  const int colBase = blockIdx.y * 64;

  v8f acc00 = {}, acc01 = {}, acc10 = {}, acc11 = {};

  const int aRow  = tid >> 1;          // 0..127
  const int aHalf = tid & 1;
  const int bK    = tid >> 3;          // 0..31
  const int bN0   = (tid & 7) * 8;

  auto stage = [&](int kk, int buf) {
    // A row-major: async global->LDS (one B128 per thread, no VGPR round trip)
    asyncCopyB128(A + (size_t)(rowBase + aRow) * K + kk + aHalf * 16,
                  &As[buf][aRow][aHalf * 16]);
    // B transposed: load 8 bf16 along N, scatter into Bs[n][k]
    uint4 bv4 = *(const uint4*)(B + (size_t)(kk + bK) * N + colBase + bN0);
    const u16* bu = (const u16*)&bv4;
    #pragma unroll
    for (int i = 0; i < 8; ++i) Bs[buf][bN0 + i][bK] = bu[i];
  };

  // preload stage 0
  stage(0, 0);
  waitAsync();
  __syncthreads();

  const int r = lane & 15, h = lane >> 4;
  int cur = 0;
  for (int kb = 0; kb < K; kb += 32) {
    int nxt = cur ^ 1;
    if (kb + 32 < K) stage(kb + 32, nxt);    // prefetch next tile into other buffer

    FragU a0, a1, b0, b1;
    {
      const u16* p0 = &As[cur][mOff + r][h * 8];
      a0.q[0] = *(const uint4*)p0;        a0.q[1] = *(const uint4*)(p0 + 16);
      const u16* p1 = &As[cur][mOff + 16 + r][h * 8];
      a1.q[0] = *(const uint4*)p1;        a1.q[1] = *(const uint4*)(p1 + 16);
      const u16* q0 = &Bs[cur][nOff + r][h * 8];
      b0.q[0] = *(const uint4*)q0;        b0.q[1] = *(const uint4*)(q0 + 16);
      const u16* q1 = &Bs[cur][nOff + 16 + r][h * 8];
      b1.q[0] = *(const uint4*)q1;        b1.q[1] = *(const uint4*)(q1 + 16);
    }
    acc00 = wmma_bf16(a0, b0, acc00);
    acc01 = wmma_bf16(a0, b1, acc01);
    acc10 = wmma_bf16(a1, b0, acc10);
    acc11 = wmma_bf16(a1, b1, acc11);

    waitAsync();
    __syncthreads();
    cur = nxt;
  }

  auto store = [&](v8f acc, int mT, int nT) {
    int n  = colBase + nT + (lane & 15);
    int mB = rowBase + mT + (lane >> 4) * 8;
    float bv = bias[n];
    #pragma unroll
    for (int r8 = 0; r8 < 8; ++r8) {
      float v = (acc[r8] + bv) * scale;
      if (RELU) v = v > 0.f ? v : 0.f;
      size_t o = (size_t)(mB + r8) * N + n;
      if (OUTBF) outB[o] = f2bf(v);
      else       outF[o] = v;
    }
  };
  store(acc00, mOff,      nOff);
  store(acc01, mOff,      nOff + 16);
  store(acc10, mOff + 16, nOff);
  store(acc11, mOff + 16, nOff + 16);
}

// ---------------- WMMA attention ----------------
// Q,K,V: bf16 [B*NPAT, D_EMB], head h at cols h*32..h*32+31; Q pre-scaled by 1/sqrt(hd).
// grid = B*H blocks, 128 threads = 4 waves. Wave handles row tiles rt = wave, wave+4, ...
// S tiles: A-frag = Q rows, B-frag = K rows (contiguous head dims, no transpose needed).
// Softmax reduced with shfl_xor 1/2/4/8 (stays within each 16-lane half = fixed rows).
// P@V via LDS: P rows (A-frag layout) + V^T (B-frag layout).
__global__ __launch_bounds__(128) void k_attn_wmma(const u16* __restrict__ Qb,
                                                   const u16* __restrict__ Kb,
                                                   const u16* __restrict__ Vb,
                                                   u16* __restrict__ O)
{
  __shared__ __align__(16) u16 Vt[HD][NKPAD];       // V^T for this (b,h): [dim][key]
  __shared__ __align__(16) u16 Pb[4][16][NKPAD];    // per-wave normalized P rows

  const int bh = blockIdx.x;
  const int b  = bh >> 4, hh = bh & 15;
  const int wave = threadIdx.x >> 5, lane = threadIdx.x & 31;
  const int l16 = lane & 15, lh = lane >> 4;
  const size_t headBase = (size_t)b * NPAT * D_EMB + (size_t)hh * HD;

  // stage V^T into LDS (coalesced global reads, transposed LDS stores); zero-pad keys>=196
  for (int key = wave; key < NKPAD; key += 4) {
    u16 val = (key < NPAT) ? Vb[headBase + (size_t)key * D_EMB + lane] : (u16)0;
    Vt[lane][key] = val;
  }
  __syncthreads();

  for (int rt = wave; rt < 13; rt += 4) {
    // ---- load Q A-fragment (rows rt*16 .. rt*16+15) ----
    int qrow = rt * 16 + l16; if (qrow > NPAT - 1) qrow = NPAT - 1;
    const u16* qp = Qb + headBase + (size_t)qrow * D_EMB + lh * 8;
    FragU qa; qa.q[0] = *(const uint4*)qp; qa.q[1] = *(const uint4*)(qp + 16);

    // ---- scores: 13 column tiles of 16 keys, one wmma each (K = hd = 32) ----
    v8f sc[13];
    float m[8];
    #pragma unroll
    for (int r = 0; r < 8; ++r) m[r] = -1e30f;
    for (int ct = 0; ct < 13; ++ct) {
      int key = ct * 16 + l16;
      int keyc = key > NPAT - 1 ? NPAT - 1 : key;
      const u16* kp = Kb + headBase + (size_t)keyc * D_EMB + lh * 8;
      FragU kf; kf.q[0] = *(const uint4*)kp; kf.q[1] = *(const uint4*)(kp + 16);
      v8f z = {};
      z = wmma_bf16(qa, kf, z);
      bool oob = key >= NPAT;
      #pragma unroll
      for (int r = 0; r < 8; ++r) {
        float vv = oob ? -1e30f : z[r];
        sc[ct][r] = vv;
        m[r] = fmaxf(m[r], vv);
      }
    }
    // row max across the 16 lanes of this half (C layout: lane=key, vgpr r=row)
    #pragma unroll
    for (int r = 0; r < 8; ++r) {
      m[r] = fmaxf(m[r], __shfl_xor(m[r], 1, 32));
      m[r] = fmaxf(m[r], __shfl_xor(m[r], 2, 32));
      m[r] = fmaxf(m[r], __shfl_xor(m[r], 4, 32));
      m[r] = fmaxf(m[r], __shfl_xor(m[r], 8, 32));
    }
    float s[8];
    #pragma unroll
    for (int r = 0; r < 8; ++r) s[r] = 0.f;
    for (int ct = 0; ct < 13; ++ct) {
      bool oob = (ct * 16 + l16) >= NPAT;
      #pragma unroll
      for (int r = 0; r < 8; ++r) {
        float e = oob ? 0.f : __expf(sc[ct][r] - m[r]);
        sc[ct][r] = e;
        s[r] += e;
      }
    }
    #pragma unroll
    for (int r = 0; r < 8; ++r) {
      s[r] += __shfl_xor(s[r], 1, 32);
      s[r] += __shfl_xor(s[r], 2, 32);
      s[r] += __shfl_xor(s[r], 4, 32);
      s[r] += __shfl_xor(s[r], 8, 32);
      s[r] = 1.f / s[r];
    }
    // ---- write normalized P (bf16) to LDS in row-major [row][key] ----
    for (int ct = 0; ct < 13; ++ct) {
      #pragma unroll
      for (int r = 0; r < 8; ++r)
        Pb[wave][r + 8 * lh][ct * 16 + l16] = f2bf(sc[ct][r] * s[r]);
    }
    #pragma unroll
    for (int r = 0; r < 8; ++r)                 // zero-pad keys 208..223
      Pb[wave][r + 8 * lh][208 + l16] = 0;
    asm volatile("s_wait_dscnt 0x0" ::: "memory");   // within-wave LDS RAW

    // ---- O = P @ V : 7 key-chunks of 32, two 16-wide dim tiles ----
    v8f o0 = {}, o1 = {};
    for (int c = 0; c < 7; ++c) {
      const u16* pp = &Pb[wave][l16][c * 32 + lh * 8];
      FragU pa; pa.q[0] = *(const uint4*)pp; pa.q[1] = *(const uint4*)(pp + 16);
      const u16* v0 = &Vt[l16][c * 32 + lh * 8];
      FragU vf0; vf0.q[0] = *(const uint4*)v0; vf0.q[1] = *(const uint4*)(v0 + 16);
      const u16* v1 = &Vt[16 + l16][c * 32 + lh * 8];
      FragU vf1; vf1.q[0] = *(const uint4*)v1; vf1.q[1] = *(const uint4*)(v1 + 16);
      o0 = wmma_bf16(pa, vf0, o0);
      o1 = wmma_bf16(pa, vf1, o1);
    }
    // ---- store O rows < 196 (C layout) ----
    #pragma unroll
    for (int r = 0; r < 8; ++r) {
      int row = rt * 16 + lh * 8 + r;
      if (row < NPAT) {
        size_t o = headBase + (size_t)row * D_EMB + l16;
        O[o]      = f2bf(o0[r]);
        O[o + 16] = f2bf(o1[r]);
      }
    }
  }
}

// ---------------- layernorm (optional residual add, optional bf16 copy) ----------------
__global__ __launch_bounds__(512) void k_ln(const float* __restrict__ a,
                                            const float* __restrict__ b,
                                            const float* __restrict__ g,
                                            const float* __restrict__ be,
                                            float* __restrict__ outF,
                                            u16* __restrict__ outB)
{
  const int t = blockIdx.x;
  const int d = threadIdx.x;
  const int lane = d & 31, wave = d >> 5;
  __shared__ float s1[16], s2[16];
  float x = a[(size_t)t * D_EMB + d];
  if (b) x += b[(size_t)t * D_EMB + d];
  float su = x, sq = x * x;
  #pragma unroll
  for (int o = 16; o > 0; o >>= 1) { su += __shfl_xor(su, o, 32); sq += __shfl_xor(sq, o, 32); }
  if (lane == 0) { s1[wave] = su; s2[wave] = sq; }
  __syncthreads();
  if (wave == 0) {
    float a1 = (lane < 16) ? s1[lane] : 0.f;
    float a2 = (lane < 16) ? s2[lane] : 0.f;
    #pragma unroll
    for (int o = 8; o > 0; o >>= 1) { a1 += __shfl_xor(a1, o, 32); a2 += __shfl_xor(a2, o, 32); }
    if (lane == 0) { s1[0] = a1; s2[0] = a2; }
  }
  __syncthreads();
  float mean = s1[0] * (1.f / D_EMB);
  float var  = s2[0] * (1.f / D_EMB) - mean * mean;
  float y = (x - mean) * rsqrtf(var + 1e-5f) * g[d] + be[d];
  if (outF) outF[(size_t)t * D_EMB + d] = y;
  if (outB) outB[(size_t)t * D_EMB + d] = f2bf(y);
}

// ---------------- host orchestration ----------------
extern "C" void kernel_launch(void* const* d_in, const int* in_sizes, int n_in,
                              void* d_out, int out_size, void* d_ws, size_t ws_size,
                              hipStream_t stream)
{
  (void)in_sizes; (void)n_in; (void)out_size; (void)ws_size;
  const float* xe    = (const float*)d_in[0];
  const int*   idx   = (const int*)  d_in[1];
  const float* mtok  = (const float*)d_in[2];
  const float* pos   = (const float*)d_in[3];
  const float* Wq    = (const float*)d_in[4];
  const float* bq    = (const float*)d_in[5];
  const float* Wk    = (const float*)d_in[6];
  const float* bk    = (const float*)d_in[7];
  const float* Wv    = (const float*)d_in[8];
  const float* bv    = (const float*)d_in[9];
  const float* Wo    = (const float*)d_in[10];
  const float* bo    = (const float*)d_in[11];
  const float* ln1g  = (const float*)d_in[12];
  const float* ln1b  = (const float*)d_in[13];
  const float* W1    = (const float*)d_in[14];
  const float* b1    = (const float*)d_in[15];
  const float* W2    = (const float*)d_in[16];
  const float* b2    = (const float*)d_in[17];
  const float* ln2g  = (const float*)d_in[18];
  const float* ln2b  = (const float*)d_in[19];
  const float* lnfg  = (const float*)d_in[20];
  const float* lnfb  = (const float*)d_in[21];

  const size_t T = T_TOK, D = D_EMB, FF = FF_DIM, L = NLAYERS;
  const float qscale = 0.17677669529663687f;   // 1/sqrt(hd)

  char* ws = (char*)d_ws;
  size_t off = 0;
  auto alloc = [&](size_t bytes) -> void* {
    void* p = ws + off;
    off += (bytes + 255) & ~(size_t)255;
    return p;
  };
  float* x   = (float*)alloc(T * D * 4);
  u16*   xb  = (u16*)  alloc(T * D * 2);
  u16*   qb  = (u16*)  alloc(T * D * 2);
  u16*   kb  = (u16*)  alloc(T * D * 2);
  u16*   vb  = (u16*)  alloc(T * D * 2);
  u16*   ob  = (u16*)  alloc(T * D * 2);
  float* att = (float*)alloc(T * D * 4);
  u16*   hb  = (u16*)  alloc(T * FF * 2);
  float* ff  = (float*)alloc(T * D * 4);
  u16* WqB = (u16*)alloc(L * D * D * 2);
  u16* WkB = (u16*)alloc(L * D * D * 2);
  u16* WvB = (u16*)alloc(L * D * D * 2);
  u16* WoB = (u16*)alloc(L * D * D * 2);
  u16* W1B = (u16*)alloc(L * D * FF * 2);
  u16* W2B = (u16*)alloc(L * FF * D * 2);

  // weight conversion to bf16 (once per launch; deterministic)
  {
    int nW = (int)(L * D * D);
    int gb = (nW + 255) / 256;
    k_f32_to_bf16<<<gb, 256, 0, stream>>>(Wq, WqB, nW);
    k_f32_to_bf16<<<gb, 256, 0, stream>>>(Wk, WkB, nW);
    k_f32_to_bf16<<<gb, 256, 0, stream>>>(Wv, WvB, nW);
    k_f32_to_bf16<<<gb, 256, 0, stream>>>(Wo, WoB, nW);
    int nF = (int)(L * D * FF);
    int gf = (nF + 255) / 256;
    k_f32_to_bf16<<<gf, 256, 0, stream>>>(W1, W1B, nF);
    k_f32_to_bf16<<<gf, 256, 0, stream>>>(W2, W2B, nF);
  }

  // embed + initial bf16 copy
  k_embed<<<(int)T, 512, 0, stream>>>(xe, idx, mtok, pos, x);
  k_f32_to_bf16<<<(int)((T * D + 255) / 256), 256, 0, stream>>>(x, xb, (int)(T * D));

  dim3 gp((unsigned)(T / 128), (unsigned)(D / 64));
  dim3 gf1((unsigned)(T / 128), (unsigned)(FF / 64));

  for (int l = 0; l < NLAYERS; ++l) {
    const u16* wq = WqB + (size_t)l * D * D;
    const u16* wk = WkB + (size_t)l * D * D;
    const u16* wv = WvB + (size_t)l * D * D;
    const u16* wo = WoB + (size_t)l * D * D;
    const u16* w1 = W1B + (size_t)l * D * FF;
    const u16* w2 = W2B + (size_t)l * FF * D;

    // Q/K/V projections straight to bf16; Q pre-scaled by 1/sqrt(hd)
    k_gemm_bf16<true, false><<<gp, 256, 0, stream>>>(xb, wq, bq + (size_t)l * D, nullptr, qb,
                                                     (int)T, (int)D, (int)D, qscale);
    k_gemm_bf16<true, false><<<gp, 256, 0, stream>>>(xb, wk, bk + (size_t)l * D, nullptr, kb,
                                                     (int)T, (int)D, (int)D, 1.0f);
    k_gemm_bf16<true, false><<<gp, 256, 0, stream>>>(xb, wv, bv + (size_t)l * D, nullptr, vb,
                                                     (int)T, (int)D, (int)D, 1.0f);

    k_attn_wmma<<<BSZ * NHEADS, 128, 0, stream>>>(qb, kb, vb, ob);

    k_gemm_bf16<false, false><<<gp, 256, 0, stream>>>(ob, wo, bo + (size_t)l * D, att, nullptr,
                                                      (int)T, (int)D, (int)D, 1.0f);
    k_ln<<<(int)T, 512, 0, stream>>>(att, x, ln1g + (size_t)l * D, ln1b + (size_t)l * D, x, xb);

    k_gemm_bf16<true, true><<<gf1, 256, 0, stream>>>(xb, w1, b1 + (size_t)l * FF, nullptr, hb,
                                                     (int)T, (int)FF, (int)D, 1.0f);
    k_gemm_bf16<false, false><<<gp, 256, 0, stream>>>(hb, w2, b2 + (size_t)l * D, ff, nullptr,
                                                      (int)T, (int)D, (int)FF, 1.0f);
    k_ln<<<(int)T, 512, 0, stream>>>(ff, x, ln2g + (size_t)l * D, ln2b + (size_t)l * D, x, xb);
  }

  // final layernorm -> d_out (f32)
  k_ln<<<(int)T, 512, 0, stream>>>(x, nullptr, lnfg, lnfb, (float*)d_out, nullptr);
}
